// Block_21646635172074
// MI455X (gfx1250) — matrix operand order repaired and checked by reading, not compiled, not run on hardware
//
#include <hip/hip_runtime.h>
#include <hip/hip_bf16.h>

// Spikformer block for MI455X (gfx1250, wave32, WMMA).
// GEMMs: v_wmma_f32_16x16x32_bf16, double-buffered LDS.
// A tile staged by the Tensor Data Mover (tensor_load_to_lds + s_wait_tensorcnt),
// falling back to global_load_async_to_lds_b128 if the TDM builtin is absent.

typedef __attribute__((ext_vector_type(16))) __bf16 v16bf;
typedef __attribute__((ext_vector_type(8)))  __bf16 v8bf;
typedef __attribute__((ext_vector_type(8)))  float  v8f;
typedef __attribute__((ext_vector_type(4)))  float  v4f;
typedef __attribute__((ext_vector_type(4)))  unsigned int v4u;
typedef __attribute__((ext_vector_type(4)))  int    v4i;
typedef __attribute__((ext_vector_type(8)))  int    v8i;

#define TT 4
#define BB 16
#define CC 256
#define NN 1024      // H*W
#define HD 1024      // 4*C
#define TB 64        // T*B
#define HEADS 8

static __device__ __forceinline__ float bf2f(__bf16 b) {
  unsigned short u = __builtin_bit_cast(unsigned short, b);
  unsigned v = ((unsigned)u) << 16;
  return __builtin_bit_cast(float, v);
}
static __device__ __forceinline__ __bf16 f2bf(float f) {
  unsigned u = __builtin_bit_cast(unsigned, f);
  unsigned r = (u + 0x7FFFu + ((u >> 16) & 1u)) >> 16;   // RNE
  return __builtin_bit_cast(__bf16, (unsigned short)r);
}

// ---- LDS byte offset (generic -> AS3 addrspacecast) -----------------------
static __device__ __forceinline__ unsigned lds_off_u32(const void* p) {
  return (unsigned)(unsigned long long)(__attribute__((address_space(3))) const void*)p;
}

// ---- CDNA5 async global->LDS (ASYNCcnt path, fallback) --------------------
static __device__ __forceinline__ void async_load_lds_b128(const __bf16* g, unsigned loff) {
  asm volatile("global_load_async_to_lds_b128 %0, %1, off"
               :: "v"(loff), "v"(g) : "memory");
}
static __device__ __forceinline__ void wait_async0() {
#if __has_builtin(__builtin_amdgcn_s_wait_asynccnt)
  __builtin_amdgcn_s_wait_asynccnt(0);
#else
  asm volatile("s_wait_asynccnt 0" ::: "memory");
#endif
}

// ---- CDNA5 Tensor Data Mover (TENSORcnt path) -----------------------------
#if __has_builtin(__builtin_amdgcn_tensor_load_to_lds)
#define HAVE_TDM 1
static __device__ __forceinline__ void wait_tensor0() {
#if __has_builtin(__builtin_amdgcn_s_wait_tensorcnt)
  __builtin_amdgcn_s_wait_tensorcnt(0);
#else
  asm volatile("s_wait_tensorcnt 0" ::: "memory");
#endif
}
// 2D tile load: tile_dim0=32 elems (64B rows), tile_dim1=64 rows,
// LDS padding 16B per 64B row -> LDS row stride 80B (40 bf16).
static __device__ __forceinline__ void tdm_load_a(const __bf16* gtile,
                                                  unsigned ldsoff, int K) {
  unsigned long long ga = (unsigned long long)(const void*)gtile;
  v4u g0;
  g0[0] = 1u;                                   // count=1, user mode
  g0[1] = ldsoff;                               // lds_addr (bytes)
  g0[2] = (unsigned)ga;                         // global_addr[31:0]
  g0[3] = (unsigned)(ga >> 32) | (2u << 30);    // addr[56:32] | type=2
  v8i g1;
  g1[0] = (int)((1u << 16)      // data_size = 2 bytes
              | (1u << 20)      // pad_enable
              | (3u << 22)      // pad_interval: 16 DWORDs (64B)
              | (3u << 25));    // pad_amount: 4 DWORDs (16B)
  g1[1] = 0;                    // tensor_dim0[15:0]=0 (dim0 = 1<<20)
  g1[2] = 16;                   // tensor_dim0[31:16]=16, tensor_dim1[15:0]=0
  g1[3] = (int)(16u | (32u << 16));  // tensor_dim1[31:16]=16, tile_dim0=32
  g1[4] = 64;                   // tile_dim1=64, tile_dim2=0
  g1[5] = K;                    // tensor_dim0_stride[31:0]
  g1[6] = 0;
  g1[7] = 0;
  v4i z4 = {};
#if defined(__clang_major__) && __clang_major__ >= 23
  v8i z8 = {};
  __builtin_amdgcn_tensor_load_to_lds(g0, g1, z4, z4, z8, 0);
#else
  __builtin_amdgcn_tensor_load_to_lds(g0, g1, z4, z4, 0);
#endif
}
#endif

// ---------------------------------------------------------------------------
// f32 -> bf16 conversion, 8 elements/thread
// ---------------------------------------------------------------------------
__global__ __launch_bounds__(256) void cvt_f32_bf16(const float* __restrict__ in,
                                                    __bf16* __restrict__ out, long n) {
  long i = ((long)blockIdx.x * 256 + threadIdx.x) * 8;
  if (i >= n) return;
  v4f a = *(const v4f*)(in + i);
  v4f b = *(const v4f*)(in + i + 4);
  v8bf o;
  #pragma unroll
  for (int e = 0; e < 4; ++e) { o[e] = f2bf(a[e]); o[e + 4] = f2bf(b[e]); }
  *(v8bf*)(out + i) = o;
}

// ---------------------------------------------------------------------------
// GEMM: Y[tb] = BN( W[M,K] @ X[tb][K,NN] (+bias) ), bf16 out.
// grid = (NN/128, M/64, TB), block = 256 (8 waves, each 16x64).
// ---------------------------------------------------------------------------
__global__ __launch_bounds__(256) void gemm_bn_kernel(
    const __bf16* __restrict__ W, const __bf16* __restrict__ X,
    __bf16* __restrict__ Y, int M, int K,
    const float* __restrict__ bias, const float* __restrict__ bn) {
  __shared__ __align__(16) __bf16 As[2][64 * 40];    // [m][k], k-padded
  __shared__ __align__(16) __bf16 BsT[2][128 * 40];  // [n][k], k-padded

  const int tb = blockIdx.z;
  const int m0 = blockIdx.y * 64;
  const int n0 = blockIdx.x * 128;
  const __bf16* Xb = X + (size_t)tb * K * NN;
  __bf16* Yb = Y + (size_t)tb * M * NN;

  const int tid  = threadIdx.x;
  const int lane = tid & 31;
  const int wave = tid >> 5;
  const int wm = wave >> 1;   // 0..3
  const int wn = wave & 1;    // 0..1
  const int l15 = lane & 15;

#if defined(HAVE_TDM)
  const unsigned aoff[2] = { lds_off_u32(&As[0][0]), lds_off_u32(&As[1][0]) };
  const __bf16* arow0 = W + (size_t)m0 * K;
#else
  const int arow = tid >> 2, acg = (tid & 3) * 8;
  const unsigned aoff[2] = { lds_off_u32(&As[0][arow * 40 + acg]),
                             lds_off_u32(&As[1][arow * 40 + acg]) };
  const __bf16* aptr = W + (size_t)(m0 + arow) * K + acg;
#endif

  // B staging role: two v8bf global loads + transposed scalar LDS stores
  const int brow = tid >> 4;            // 0..15 (k), +16 for j=1
  const int bcg  = (tid & 15) * 8;      // n group
  const __bf16* bptr = Xb + (size_t)brow * NN + n0 + bcg;

  v8f acc[4] = {{}, {}, {}, {}};
  const int nk = K / 32;

  // ---- prologue: stage tile 0 into buffer 0 ----
#if defined(HAVE_TDM)
  if (wave == 0) tdm_load_a(arow0, aoff[0], K);
#else
  async_load_lds_b128(aptr, aoff[0]);
#endif
  {
    v8bf b0 = *(const v8bf*)bptr;
    v8bf b1 = *(const v8bf*)(bptr + (size_t)16 * NN);
    #pragma unroll
    for (int e = 0; e < 8; ++e) {
      BsT[0][(bcg + e) * 40 + brow] = b0[e];
      BsT[0][(bcg + e) * 40 + brow + 16] = b1[e];
    }
  }
#if defined(HAVE_TDM)
  if (wave == 0) wait_tensor0();
#else
  wait_async0();
#endif
  __syncthreads();

  for (int k = 0; k < nk; ++k) {
    const int cur = k & 1, nxt = cur ^ 1;
    const bool more = (k + 1 < nk);
    v8bf b0, b1;
    if (more) {
      const int kt = (k + 1) * 32;
#if defined(HAVE_TDM)
      if (wave == 0) tdm_load_a(arow0 + kt, aoff[nxt], K);
#else
      async_load_lds_b128(aptr + kt, aoff[nxt]);
#endif
      b0 = *(const v8bf*)(bptr + (size_t)kt * NN);
      b1 = *(const v8bf*)(bptr + (size_t)(kt + 16) * NN);
    }

    // A fragment (16x32 bf16, ISA 7.12.2 layout)
    v16bf af;
    {
      int m = wm * 16 + l15;
      int klo = (lane < 16) ? 0 : 8;
      v8bf lo = *(const v8bf*)&As[cur][m * 40 + klo];
      v8bf hi = *(const v8bf*)&As[cur][m * 40 + klo + 16];
      #pragma unroll
      for (int i = 0; i < 8; ++i) { af[i] = lo[i]; af[i + 8] = hi[i]; }
    }
    // batch all four B fragments, then four back-to-back WMMAs
    const int kb = (lane < 16) ? 0 : 8;
    v16bf bfr[4];
    #pragma unroll
    for (int sub = 0; sub < 4; ++sub) {
      int n = wn * 64 + sub * 16 + l15;
      v8bf lo = *(const v8bf*)&BsT[cur][n * 40 + kb];
      v8bf hi = *(const v8bf*)&BsT[cur][n * 40 + kb + 16];
      #pragma unroll
      for (int i = 0; i < 8; ++i) { bfr[sub][i] = lo[i]; bfr[sub][i + 8] = hi[i]; }
    }
    #pragma unroll
    for (int sub = 0; sub < 4; ++sub)
      acc[sub] = __builtin_amdgcn_wmma_f32_16x16x32_bf16(false, af, false, bfr[sub],
                                                         (short)0, acc[sub],
                                                         false, false);

    if (more) {
      #pragma unroll
      for (int e = 0; e < 8; ++e) {
        BsT[nxt][(bcg + e) * 40 + brow] = b0[e];
        BsT[nxt][(bcg + e) * 40 + brow + 16] = b1[e];
      }
    }
#if defined(HAVE_TDM)
    if (more && wave == 0) wait_tensor0();
#else
    wait_async0();
#endif
    __syncthreads();
  }

  // epilogue: bias + eval-mode BN, bf16 store
  #pragma unroll
  for (int sub = 0; sub < 4; ++sub) {
    int n = n0 + wn * 64 + sub * 16 + l15;
    #pragma unroll
    for (int r = 0; r < 8; ++r) {
      int m = m0 + wm * 16 + r + ((lane < 16) ? 0 : 8);
      float y = acc[sub][r];
      if (bias) y += bias[m];
      float g = bn[0 * M + m], be = bn[1 * M + m];
      float mu = bn[2 * M + m], va = bn[3 * M + m];
      y = g * (y - mu) * rsqrtf(va + 1e-5f) + be;
      Yb[(size_t)m * NN + n] = f2bf(y);
    }
  }
}

// ---------------------------------------------------------------------------
// Linear attention per (t,b,head):
//   KV = Kh @ Vh^T (32x32, K=1024) -> scaled KV^T in LDS
//   Ah = (scale*KV)^T @ Qh         (32x1024, K=32)
// grid = TB*HEADS, block = 256
// ---------------------------------------------------------------------------
__global__ __launch_bounds__(256) void attn_kernel(
    const __bf16* __restrict__ Q, const __bf16* __restrict__ Kb,
    const __bf16* __restrict__ V, __bf16* __restrict__ A) {
  __shared__ __align__(16) __bf16 kvT[32 * 40];  // [e][d]

  const int id = blockIdx.x;
  const int h  = id & (HEADS - 1);
  const int tb = id / HEADS;
  const size_t base = ((size_t)tb * CC + h * 32) * NN;
  const __bf16* qh = Q + base;
  const __bf16* kh = Kb + base;
  const __bf16* vh = V + base;
  __bf16* ah = A + base;

  const int tid = threadIdx.x;
  const int lane = tid & 31;
  const int wave = tid >> 5;
  const int l15 = lane & 15;

  if (wave < 4) {                       // phase 1: KV, one 16x16 tile per wave
    const int mt = (wave >> 1) * 16;    // d tile
    const int nt = (wave & 1) * 16;     // e tile
    v8f c = {};
    const int klo = (lane < 16) ? 0 : 8;
    const int kb  = (lane < 16) ? 0 : 8;
    for (int kt = 0; kt < NN; kt += 32) {
      v16bf af, bf;
      {
        const __bf16* p = kh + (size_t)(mt + l15) * NN + kt + klo;
        v8bf lo = *(const v8bf*)p;
        v8bf hi = *(const v8bf*)(p + 16);
        #pragma unroll
        for (int i = 0; i < 8; ++i) { af[i] = lo[i]; af[i + 8] = hi[i]; }
      }
      {
        const __bf16* p = vh + (size_t)(nt + l15) * NN + kt + kb;
        v8bf lo = *(const v8bf*)p;
        v8bf hi = *(const v8bf*)(p + 16);
        #pragma unroll
        for (int i = 0; i < 8; ++i) { bf[i] = lo[i]; bf[i + 8] = hi[i]; }
      }
      c = __builtin_amdgcn_wmma_f32_16x16x32_bf16(false, af, false, bf,
                                                  (short)0, c, false, false);
    }
    #pragma unroll
    for (int r = 0; r < 8; ++r) {
      int d = mt + r + ((lane < 16) ? 0 : 8);
      int e = nt + l15;
      kvT[e * 40 + d] = f2bf(0.125f * c[r]);   // fold scale
    }
  }
  __syncthreads();

  // phase 2: Ah = kvT @ Qh, each wave covers 128 columns
  v16bf a0, a1;
  {
    const int klo = (lane < 16) ? 0 : 16;
    #pragma unroll
    for (int et = 0; et < 2; ++et) {
      int m = et * 16 + l15;
      v8bf lo = *(const v8bf*)&kvT[m * 40 + klo];
      v8bf hi = *(const v8bf*)&kvT[m * 40 + klo + 8];
      v16bf f;
      #pragma unroll
      for (int i = 0; i < 8; ++i) { f[i] = lo[i]; f[i + 8] = hi[i]; }
      if (et == 0) a0 = f; else a1 = f;
    }
  }
  const int nbase = wave * 128;
  const int kb = (lane < 16) ? 0 : 16;
  for (int nt = 0; nt < 8; ++nt) {
    int n = nbase + nt * 16 + l15;
    v16bf bf;
    #pragma unroll
    for (int idx = 0; idx < 16; ++idx)
      bf[idx] = qh[(size_t)(kb + idx) * NN + n];
    #pragma unroll
    for (int et = 0; et < 2; ++et) {
      v8f c = {};
      c = __builtin_amdgcn_wmma_f32_16x16x32_bf16(false, (et == 0) ? a0 : a1,
                                                  false, bf, (short)0, c,
                                                  false, false);
      #pragma unroll
      for (int r = 0; r < 8; ++r) {
        int e = et * 16 + r + ((lane < 16) ? 0 : 8);
        ah[(size_t)e * NN + n] = f2bf(c[r]);
      }
    }
  }
}

// ---------------------------------------------------------------------------
// In-place multi-step LIF scan over T=4 (tau=2, hard reset), 8 elems/thread.
// ---------------------------------------------------------------------------
__global__ __launch_bounds__(256) void lif_kernel(__bf16* __restrict__ Y, long S,
                                                  float vth) {
  long i = ((long)blockIdx.x * 256 + threadIdx.x) * 8;
  if (i >= S) return;
  float v[8] = {};
  #pragma unroll
  for (int t = 0; t < TT; ++t) {
    v8bf d = *(const v8bf*)(Y + (long)t * S + i);
    v8bf s;
    #pragma unroll
    for (int e = 0; e < 8; ++e) {
      float xx = bf2f(d[e]);
      v[e] += (xx - v[e]) * 0.5f;
      float sp = (v[e] >= vth) ? 1.f : 0.f;
      v[e] *= (1.f - sp);
      s[e] = f2bf(sp);
    }
    *(v8bf*)(Y + (long)t * S + i) = s;
  }
}

// x1 = bf16(x + attn_spikes), 8 elems/thread
__global__ __launch_bounds__(256) void residual_kernel(const float* __restrict__ x,
                                                       const __bf16* __restrict__ s,
                                                       __bf16* __restrict__ o, long n) {
  long i = ((long)blockIdx.x * 256 + threadIdx.x) * 8;
  if (i >= n) return;
  v4f a = *(const v4f*)(x + i);
  v4f b = *(const v4f*)(x + i + 4);
  v8bf sp = *(const v8bf*)(s + i);
  v8bf o8;
  #pragma unroll
  for (int e = 0; e < 4; ++e) {
    o8[e]     = f2bf(a[e] + bf2f(sp[e]));
    o8[e + 4] = f2bf(b[e] + bf2f(sp[e + 4]));
  }
  *(v8bf*)(o + i) = o8;
}

// out = x + attn_spikes + mlp_spikes (f32), 8 elems/thread
__global__ __launch_bounds__(256) void final_kernel(const float* __restrict__ x,
                                                    const __bf16* __restrict__ sa,
                                                    const __bf16* __restrict__ sm,
                                                    float* __restrict__ o, long n) {
  long i = ((long)blockIdx.x * 256 + threadIdx.x) * 8;
  if (i >= n) return;
  v4f a = *(const v4f*)(x + i);
  v4f b = *(const v4f*)(x + i + 4);
  v8bf p = *(const v8bf*)(sa + i);
  v8bf q = *(const v8bf*)(sm + i);
  v4f o0, o1;
  #pragma unroll
  for (int e = 0; e < 4; ++e) {
    o0[e] = a[e] + bf2f(p[e]) + bf2f(q[e]);
    o1[e] = b[e] + bf2f(p[e + 4]) + bf2f(q[e + 4]);
  }
  *(v4f*)(o + i) = o0;
  *(v4f*)(o + i + 4) = o1;
}

// ---------------------------------------------------------------------------
extern "C" void kernel_launch(void* const* d_in, const int* in_sizes, int n_in,
                              void* d_out, int out_size, void* d_ws, size_t ws_size,
                              hipStream_t stream) {
  const float* x      = (const float*)d_in[0];
  const float* wq     = (const float*)d_in[2];
  const float* bn_q   = (const float*)d_in[3];
  const float* wk     = (const float*)d_in[4];
  const float* bn_k   = (const float*)d_in[5];
  const float* wv     = (const float*)d_in[6];
  const float* bn_v   = (const float*)d_in[7];
  const float* w_proj = (const float*)d_in[8];
  const float* b_proj = (const float*)d_in[9];
  const float* bn_pr  = (const float*)d_in[10];
  const float* w_fc1  = (const float*)d_in[11];
  const float* b_fc1  = (const float*)d_in[12];
  const float* bn_f1  = (const float*)d_in[13];
  const float* w_fc2  = (const float*)d_in[14];
  const float* b_fc2  = (const float*)d_in[15];
  const float* bn_f2  = (const float*)d_in[16];
  float* out = (float*)d_out;

  const long TBCN = (long)TB * CC * NN;          // 16,777,216
  const long TBHN = (long)TB * HD * NN;          // 67,108,864
  char* ws = (char*)d_ws;
  size_t off = 0;
  __bf16* xb = (__bf16*)(ws + off); off += (size_t)TBCN * 2;
  __bf16* qb = (__bf16*)(ws + off); off += (size_t)TBCN * 2;   // later: attn spikes
  __bf16* kb = (__bf16*)(ws + off); off += (size_t)TBCN * 2;   // later: x1
  __bf16* vb = (__bf16*)(ws + off); off += (size_t)TBCN * 2;   // later: mlp spikes
  __bf16* ab = (__bf16*)(ws + off); off += (size_t)TBCN * 2;
  __bf16* h1 = (__bf16*)(ws + off); off += (size_t)TBHN * 2;
  __bf16* wqb = (__bf16*)(ws + off); off += (size_t)CC * CC * 2;
  __bf16* wkb = (__bf16*)(ws + off); off += (size_t)CC * CC * 2;
  __bf16* wvb = (__bf16*)(ws + off); off += (size_t)CC * CC * 2;
  __bf16* wpb = (__bf16*)(ws + off); off += (size_t)CC * CC * 2;
  __bf16* w1b = (__bf16*)(ws + off); off += (size_t)HD * CC * 2;
  __bf16* w2b = (__bf16*)(ws + off); off += (size_t)CC * HD * 2;

  const long SC = (long)BB * CC * NN;   // per-t elements, C channels
  const long SH = (long)BB * HD * NN;   // per-t elements, Hd channels

  dim3 blk(256);
  cvt_f32_bf16<<<(CC * CC) / 2048, blk, 0, stream>>>(wq, wqb, (long)CC * CC);
  cvt_f32_bf16<<<(CC * CC) / 2048, blk, 0, stream>>>(wk, wkb, (long)CC * CC);
  cvt_f32_bf16<<<(CC * CC) / 2048, blk, 0, stream>>>(wv, wvb, (long)CC * CC);
  cvt_f32_bf16<<<(CC * CC) / 2048, blk, 0, stream>>>(w_proj, wpb, (long)CC * CC);
  cvt_f32_bf16<<<(HD * CC) / 2048, blk, 0, stream>>>(w_fc1, w1b, (long)HD * CC);
  cvt_f32_bf16<<<(CC * HD) / 2048, blk, 0, stream>>>(w_fc2, w2b, (long)CC * HD);
  cvt_f32_bf16<<<(int)(TBCN / 2048), blk, 0, stream>>>(x, xb, TBCN);

  dim3 gC(NN / 128, CC / 64, TB);   // M=256
  dim3 gH(NN / 128, HD / 64, TB);   // M=1024

  // q/k/v projections + BN + LIF
  gemm_bn_kernel<<<gC, blk, 0, stream>>>(wqb, xb, qb, CC, CC, nullptr, bn_q);
  gemm_bn_kernel<<<gC, blk, 0, stream>>>(wkb, xb, kb, CC, CC, nullptr, bn_k);
  gemm_bn_kernel<<<gC, blk, 0, stream>>>(wvb, xb, vb, CC, CC, nullptr, bn_v);
  lif_kernel<<<(int)(SC / 2048), blk, 0, stream>>>(qb, SC, 1.0f);
  lif_kernel<<<(int)(SC / 2048), blk, 0, stream>>>(kb, SC, 1.0f);
  lif_kernel<<<(int)(SC / 2048), blk, 0, stream>>>(vb, SC, 1.0f);

  // linear attention + attn LIF (v_th = 0.5)
  attn_kernel<<<TB * HEADS, blk, 0, stream>>>(qb, kb, vb, ab);
  lif_kernel<<<(int)(SC / 2048), blk, 0, stream>>>(ab, SC, 0.5f);

  // proj + BN + LIF -> attn spikes (reuse qb)
  gemm_bn_kernel<<<gC, blk, 0, stream>>>(wpb, ab, qb, CC, CC, b_proj, bn_pr);
  lif_kernel<<<(int)(SC / 2048), blk, 0, stream>>>(qb, SC, 1.0f);

  // x1 = x + attn_out (bf16, reuse kb)
  residual_kernel<<<(int)(TBCN / 2048), blk, 0, stream>>>(x, qb, kb, TBCN);

  // MLP
  gemm_bn_kernel<<<gH, blk, 0, stream>>>(w1b, kb, h1, HD, CC, b_fc1, bn_f1);
  lif_kernel<<<(int)(SH / 2048), blk, 0, stream>>>(h1, SH, 1.0f);
  gemm_bn_kernel<<<gC, blk, 0, stream>>>(w2b, h1, vb, CC, HD, b_fc2, bn_f2);
  lif_kernel<<<(int)(SC / 2048), blk, 0, stream>>>(vb, SC, 1.0f);

  // out = x + attn_spikes + mlp_spikes
  final_kernel<<<(int)(TBCN / 2048), blk, 0, stream>>>(x, qb, vb, out, TBCN);
}
